// VarAttention_55104430407908
// MI455X (gfx1250) — matrix operand order repaired and checked by reading, not compiled, not run on hardware
//
#include <hip/hip_runtime.h>
#include <hip/hip_bf16.h>

typedef __bf16 bf16_t;
typedef __attribute__((ext_vector_type(16))) __bf16 v16bf;
typedef __attribute__((ext_vector_type(8)))  float  v8f;

#define BTOT 8
#define NTOK 3137
#define DIMM 768
#define HH   12
#define HD   64
#define FF   16
#define NSEG 196
#define MTOT (BTOT*NTOK)   /* 25096 */
#define QKVN (3*DIMM)      /* 2304  */

// async-to-LDS builtin signature (from hipcc diagnostic): (v4i AS1*, v4i AS3*, Imm, Imm)
typedef int v4i_t __attribute__((vector_size(16)));
typedef __attribute__((address_space(1))) v4i_t as1_v4i;
typedef __attribute__((address_space(3))) v4i_t as3_v4i;

#if defined(__AMDGCN__) && __has_builtin(__builtin_amdgcn_global_load_async_to_lds_b128)
#define USE_ASYNC 1
#else
#define USE_ASYNC 0
#endif

__device__ __forceinline__ bf16_t f2bf(float f) {
  unsigned u = __builtin_bit_cast(unsigned, f);
  unsigned r = u + 0x7FFFu + ((u >> 16) & 1u);   // round-to-nearest-even
  unsigned short h = (unsigned short)(r >> 16);
  return __builtin_bit_cast(bf16_t, h);
}
__device__ __forceinline__ float bf2f(bf16_t b) {
  unsigned short h = __builtin_bit_cast(unsigned short, b);
  return __builtin_bit_cast(float, ((unsigned)h) << 16);
}
// pack two f32 -> two bf16 (v_cvt_pk_bf16_f32 when available)
__device__ __forceinline__ void cvt2(bf16_t* dst, float x, float y) {
#if defined(__AMDGCN__) && __has_builtin(__builtin_amdgcn_cvt_pk_bf16_f32)
  auto p = __builtin_amdgcn_cvt_pk_bf16_f32(x, y);
  __builtin_memcpy((void*)dst, &p, 4);
#else
  dst[0] = f2bf(x); dst[1] = f2bf(y);
#endif
}
__device__ __forceinline__ v8f zero8() {
  v8f z;
#pragma unroll
  for (int i = 0; i < 8; ++i) z[i] = 0.f;
  return z;
}

#if USE_ASYNC
__device__ __forceinline__ void async_copy16(const bf16_t* g, bf16_t* l) {
  __builtin_amdgcn_global_load_async_to_lds_b128(
      (as1_v4i*)const_cast<bf16_t*>(g), (as3_v4i*)l, 0, 0);
}
__device__ __forceinline__ void wait_async0() {
#if __has_builtin(__builtin_amdgcn_s_wait_asynccnt)
  __builtin_amdgcn_s_wait_asynccnt(0);
#else
  asm volatile("s_wait_asynccnt 0x0" ::: "memory");
#endif
}
#endif

// ---------------------------------------------------------------------------
// Kernel 1: qkv = x @ Wqkv^T  (f32 in, bf16 out). 128x64 C tile per block,
// 4 waves x 32 rows each, K-step 32, 8 WMMA per wave per k-step.
// ---------------------------------------------------------------------------
__global__ __launch_bounds__(128) void gemm_qkv(const float* __restrict__ X,
                                                const float* __restrict__ W,
                                                bf16_t* __restrict__ Out) {
  __shared__ __align__(16) bf16_t Asm[128 * 40];
  __shared__ __align__(16) bf16_t Bsm[64 * 40];
  const int tid = threadIdx.x, lane = tid & 31, wid = tid >> 5;
  const int half = lane >> 4, lrow = lane & 15;
  const int n0 = blockIdx.x * 64;
  const int m0 = blockIdx.y * 128;
  const int r0 = wid * 32;
  v8f acc[8];
#pragma unroll
  for (int i = 0; i < 8; ++i) acc[i] = zero8();

  for (int kt = 0; kt < 24; ++kt) {
    const int k0 = kt * 32;
#pragma unroll
    for (int i = 0; i < 8; ++i) {          // A tile: 128x32 f32 -> bf16
      int e = tid + i * 128;
      int row = e >> 3, c4 = e & 7;
      int gm = m0 + row;
      float4 va = make_float4(0.f, 0.f, 0.f, 0.f);
      if (gm < MTOT) va = *(const float4*)(X + (size_t)gm * DIMM + k0 + c4 * 4);
      bf16_t* dA = &Asm[row * 40 + c4 * 4];
      cvt2(dA, va.x, va.y); cvt2(dA + 2, va.z, va.w);
    }
#pragma unroll
    for (int i = 0; i < 4; ++i) {          // B tile: 64x32 f32 -> bf16
      int e = tid + i * 128;
      int row = e >> 3, c4 = e & 7;
      float4 vb = *(const float4*)(W + (size_t)(n0 + row) * DIMM + k0 + c4 * 4);
      bf16_t* dB = &Bsm[row * 40 + c4 * 4];
      cvt2(dB, vb.x, vb.y); cvt2(dB + 2, vb.z, vb.w);
    }
    __syncthreads();
    if (kt + 1 < 24) {                     // global_prefetch_b8 of next tiles
      int gm = m0 + tid; if (gm >= MTOT) gm = MTOT - 1;
      __builtin_prefetch(X + (size_t)gm * DIMM + k0 + 32, 0, 1);
      __builtin_prefetch(W + (size_t)(n0 + (tid >> 1)) * DIMM + k0 + 32 + (tid & 1) * 16, 0, 1);
    }
    v16bf af0, af1;                        // A frags for rows r0.. and r0+16..
#pragma unroll
    for (int e = 0; e < 16; ++e) {
      int kk = e + 8 * half + ((e >= 8) ? 8 : 0);
      af0[e] = Asm[(r0 + lrow) * 40 + kk];
      af1[e] = Asm[(r0 + 16 + lrow) * 40 + kk];
    }
#pragma unroll
    for (int j = 0; j < 4; ++j) {
      v16bf bfv;                           // B frag: K = e + 8*half (+8 for e>=8), N = lane&15
#pragma unroll
      for (int e = 0; e < 16; ++e) bfv[e] = Bsm[(j * 16 + lrow) * 40 + e + 8 * half + ((e >= 8) ? 8 : 0)];
      acc[j]     = __builtin_amdgcn_wmma_f32_16x16x32_bf16(false, af0, false, bfv,
                                                           (short)0, acc[j], false, false);
      acc[4 + j] = __builtin_amdgcn_wmma_f32_16x16x32_bf16(false, af1, false, bfv,
                                                           (short)0, acc[4 + j], false, false);
    }
    __syncthreads();
  }
#pragma unroll
  for (int j = 0; j < 4; ++j)
#pragma unroll
    for (int v = 0; v < 8; ++v) {
      int gm0 = m0 + r0 + v + 8 * half;
      int gm1 = gm0 + 16;
      int col = n0 + j * 16 + lrow;
      if (gm0 < MTOT) Out[(size_t)gm0 * QKVN + col] = f2bf(acc[j][v]);
      if (gm1 < MTOT) Out[(size_t)gm1 * QKVN + col] = f2bf(acc[4 + j][v]);
    }
}

// ---------------------------------------------------------------------------
// Kernel 2: segmented attention. Block = (segment, 16-query M-tile), 4 waves.
// ---------------------------------------------------------------------------
__global__ __launch_bounds__(128) void seg_attn(const bf16_t* __restrict__ qkv,
                                                bf16_t* __restrict__ attn) {
  __shared__ __align__(16) bf16_t KV[224 * 64];  // K tiles then V tiles
  __shared__ __align__(16) bf16_t Qs[16 * 64];
  __shared__ float Ss[16 * 224];                 // S then P (f32)
  const int seg = blockIdx.x, mtile = blockIdx.y;
  const int bh = seg >> 4, sIdx = seg & 15;
  const int b = bh / HH, h = bh % HH;
  const int tid = threadIdx.x, lane = tid & 31, wid = tid >> 5;
  const int half = lane >> 4, lrow = lane & 15;
  const size_t baseTok = (size_t)b * NTOK;

  // ---- stage Q (16x64) + K (208x64) via async-to-LDS when available
  {
    int row = tid >> 3, c = tid & 7;
    int ql = mtile * 16 + row;
    if (ql < NSEG) {
      const bf16_t* src = qkv + (baseTok + 1 + (size_t)sIdx * NSEG + ql) * QKVN + h * HD + c * 8;
#if USE_ASYNC
      async_copy16(src, &Qs[row * 64 + c * 8]);
#else
      *(uint4*)(&Qs[row * 64 + c * 8]) = *(const uint4*)src;
#endif
    } else {
      *(uint4*)(&Qs[row * 64 + c * 8]) = make_uint4(0u, 0u, 0u, 0u);
    }
  }
#pragma unroll
  for (int i = 0; i < 13; ++i) {
    int e = tid + i * 128;
    int row = e >> 3, c = e & 7;
    if (row < NSEG + 1) {
      size_t t = baseTok + ((row == 0) ? 0 : (1 + (size_t)sIdx * NSEG + (row - 1)));
      const bf16_t* src = qkv + t * QKVN + DIMM + h * HD + c * 8;
#if USE_ASYNC
      async_copy16(src, &KV[row * 64 + c * 8]);
#else
      *(uint4*)(&KV[row * 64 + c * 8]) = *(const uint4*)src;
#endif
    } else {
      *(uint4*)(&KV[row * 64 + c * 8]) = make_uint4(0u, 0u, 0u, 0u);
    }
  }
#if USE_ASYNC
  wait_async0();
#endif
  __syncthreads();

  // ---- S = (Q K^T) * scale : 13 key tiles over 4 waves, 2 WMMA each (K=64)
  v16bf a0, a1;
#pragma unroll
  for (int e = 0; e < 16; ++e) {
    int kk = e + 8 * half + ((e >= 8) ? 8 : 0);
    a0[e] = Qs[lrow * 64 + kk];
    a1[e] = Qs[lrow * 64 + 32 + kk];
  }
  for (int nt = wid; nt < 13; nt += 4) {
    v16bf b0, b1;
    const int kr = (nt * 16 + lrow) * 64;
#pragma unroll
    for (int e = 0; e < 16; ++e) {
      b0[e] = KV[kr + e + 8 * half + ((e >= 8) ? 8 : 0)];
      b1[e] = KV[kr + 32 + e + 8 * half + ((e >= 8) ? 8 : 0)];
    }
    v8f s = zero8();
    s = __builtin_amdgcn_wmma_f32_16x16x32_bf16(false, a0, false, b0, (short)0, s, false, false);
    s = __builtin_amdgcn_wmma_f32_16x16x32_bf16(false, a1, false, b1, (short)0, s, false, false);
#pragma unroll
    for (int v = 0; v < 8; ++v)
      Ss[(v + 8 * half) * 224 + nt * 16 + lrow] = s[v] * 0.125f;  // 1/sqrt(64)
  }
  __syncthreads();

  // ---- softmax over 197 valid cols; 8 threads per row
  {
    const int row = tid >> 3, sub = tid & 7;
    float* Sr = &Ss[row * 224];
    float mx = -1e30f;
    for (int c = sub; c < 197; c += 8) mx = fmaxf(mx, Sr[c]);
#pragma unroll
    for (int o = 4; o >= 1; o >>= 1) mx = fmaxf(mx, __shfl_xor(mx, o, 8));
    float sm = 0.f;
    for (int c = sub; c < 197; c += 8) { float ev = __expf(Sr[c] - mx); Sr[c] = ev; sm += ev; }
#pragma unroll
    for (int o = 4; o >= 1; o >>= 1) sm += __shfl_xor(sm, o, 8);
    float inv = 1.0f / sm;
    for (int c = sub; c < 197; c += 8) Sr[c] *= inv;
    for (int c = 197 + sub; c < 224; c += 8) Sr[c] = 0.f;  // zero K-padding of P
  }
  __syncthreads();

  // ---- stage V (224x64), zero-padded past row 196
#pragma unroll
  for (int i = 0; i < 14; ++i) {
    int e = tid + i * 128;
    int row = e >> 3, c = e & 7;
    if (row < NSEG + 1) {
      size_t t = baseTok + ((row == 0) ? 0 : (1 + (size_t)sIdx * NSEG + (row - 1)));
      const bf16_t* src = qkv + t * QKVN + 2 * DIMM + h * HD + c * 8;
#if USE_ASYNC
      async_copy16(src, &KV[row * 64 + c * 8]);
#else
      *(uint4*)(&KV[row * 64 + c * 8]) = *(const uint4*)src;
#endif
    } else {
      *(uint4*)(&KV[row * 64 + c * 8]) = make_uint4(0u, 0u, 0u, 0u);
    }
  }
#if USE_ASYNC
  wait_async0();
#endif
  __syncthreads();

  // ---- O = P V : each wave owns one 16-col output tile, 7 K-steps of 32
  const int c0 = wid * 16;
  v8f o = zero8();
  for (int kb = 0; kb < 7; ++kb) {
    v16bf pa, vb;
#pragma unroll
    for (int e = 0; e < 16; ++e) {
      int pc = kb * 32 + e + 8 * half + ((e >= 8) ? 8 : 0);
      pa[e] = f2bf(Ss[lrow * 224 + pc]);
      vb[e] = KV[(kb * 32 + e + 8 * half + ((e >= 8) ? 8 : 0)) * 64 + c0 + lrow];
    }
    o = __builtin_amdgcn_wmma_f32_16x16x32_bf16(false, pa, false, vb, (short)0, o, false, false);
  }
#pragma unroll
  for (int v = 0; v < 8; ++v) {
    int ql = mtile * 16 + v + 8 * half;
    if (ql < NSEG) {
      size_t t = baseTok + 1 + (size_t)sIdx * NSEG + ql;
      attn[t * DIMM + h * HD + c0 + lrow] = f2bf(o[v]);
    }
  }
}

// ---------------------------------------------------------------------------
// Kernel 3: cls-token attention over all 3137 keys (tiny; VALU)
// ---------------------------------------------------------------------------
__global__ __launch_bounds__(256) void cls_attn(const bf16_t* __restrict__ qkv,
                                                bf16_t* __restrict__ attn) {
  __shared__ float qsh[64];
  __shared__ float logits[NTOK];
  __shared__ float redm[8];
  __shared__ float reds[8];
  const int b = blockIdx.x / HH, h = blockIdx.x % HH;
  const int tid = threadIdx.x;
  const size_t base = (size_t)b * NTOK;
  if (tid < 64) qsh[tid] = bf2f(qkv[base * QKVN + h * HD + tid]) * 0.125f;
  __syncthreads();

  float lmax = -1e30f;
  for (int j = tid; j < NTOK; j += 256) {
    const bf16_t* kp = qkv + (base + j) * QKVN + DIMM + h * HD;
    float d = 0.f;
#pragma unroll
    for (int u = 0; u < HD; ++u) d += qsh[u] * bf2f(kp[u]);
    logits[j] = d;
    lmax = fmaxf(lmax, d);
  }
#pragma unroll
  for (int o = 16; o >= 1; o >>= 1) lmax = fmaxf(lmax, __shfl_xor(lmax, o, 32));
  if ((tid & 31) == 0) redm[tid >> 5] = lmax;
  __syncthreads();
  if (tid < 8) {
    float m = redm[tid];
#pragma unroll
    for (int o = 4; o >= 1; o >>= 1) m = fmaxf(m, __shfl_xor(m, o, 8));
    if (tid == 0) redm[0] = m;
  }
  __syncthreads();
  const float mx = redm[0];

  float lsum = 0.f;
  for (int j = tid; j < NTOK; j += 256) {
    float ev = __expf(logits[j] - mx);
    logits[j] = ev;
    lsum += ev;
  }
#pragma unroll
  for (int o = 16; o >= 1; o >>= 1) lsum += __shfl_xor(lsum, o, 32);
  if ((tid & 31) == 0) reds[tid >> 5] = lsum;
  __syncthreads();
  if (tid < 8) {
    float s = reds[tid];
#pragma unroll
    for (int o = 4; o >= 1; o >>= 1) s += __shfl_xor(s, o, 8);
    if (tid == 0) reds[0] = s;
  }
  __syncthreads();
  const float inv = 1.0f / reds[0];

  const int d = tid >> 2, g = tid & 3;   // 4 threads per head-dim
  float accv = 0.f;
  for (int j = g; j < NTOK; j += 4)
    accv += logits[j] * bf2f(qkv[(base + j) * QKVN + 2 * DIMM + h * HD + d]);
  accv += __shfl_xor(accv, 1, 4);
  accv += __shfl_xor(accv, 2, 4);
  if (g == 0) attn[base * DIMM + h * HD + d] = f2bf(accv * inv);
}

// ---------------------------------------------------------------------------
// Kernel 4: out = attn @ Wproj^T + bproj  (bf16 A async-staged, f32 out)
// ---------------------------------------------------------------------------
__global__ __launch_bounds__(128) void gemm_proj(const bf16_t* __restrict__ A,
                                                 const float* __restrict__ W,
                                                 const float* __restrict__ bias,
                                                 float* __restrict__ Out) {
  __shared__ __align__(16) bf16_t Asm[128 * 40];
  __shared__ __align__(16) bf16_t Bsm[64 * 40];
  const int tid = threadIdx.x, lane = tid & 31, wid = tid >> 5;
  const int half = lane >> 4, lrow = lane & 15;
  const int n0 = blockIdx.x * 64;
  const int m0 = blockIdx.y * 128;
  const int r0 = wid * 32;
  v8f acc[8];
#pragma unroll
  for (int i = 0; i < 8; ++i) acc[i] = zero8();

  for (int kt = 0; kt < 24; ++kt) {
    const int k0 = kt * 32;
#pragma unroll
    for (int i = 0; i < 4; ++i) {          // A tile: 128x32 bf16 (16B copies)
      int e = tid + i * 128;
      int row = e >> 2, c8 = e & 3;
      int gm = m0 + row;
      if (gm < MTOT) {
        const bf16_t* src = A + (size_t)gm * DIMM + k0 + c8 * 8;
#if USE_ASYNC
        async_copy16(src, &Asm[row * 40 + c8 * 8]);
#else
        *(uint4*)(&Asm[row * 40 + c8 * 8]) = *(const uint4*)src;
#endif
      } else {
        *(uint4*)(&Asm[row * 40 + c8 * 8]) = make_uint4(0u, 0u, 0u, 0u);
      }
    }
#pragma unroll
    for (int i = 0; i < 4; ++i) {          // W tile: 64x32 f32 -> bf16
      int e = tid + i * 128;
      int row = e >> 3, c4 = e & 7;
      float4 vb = *(const float4*)(W + (size_t)(n0 + row) * DIMM + k0 + c4 * 4);
      bf16_t* dB = &Bsm[row * 40 + c4 * 4];
      cvt2(dB, vb.x, vb.y); cvt2(dB + 2, vb.z, vb.w);
    }
#if USE_ASYNC
    wait_async0();
#endif
    __syncthreads();
    if (kt + 1 < 24) {
      int gm = m0 + tid; if (gm >= MTOT) gm = MTOT - 1;
      __builtin_prefetch(A + (size_t)gm * DIMM + k0 + 32, 0, 1);
      __builtin_prefetch(W + (size_t)(n0 + (tid >> 1)) * DIMM + k0 + 32 + (tid & 1) * 16, 0, 1);
    }
    v16bf af0, af1;
#pragma unroll
    for (int e = 0; e < 16; ++e) {
      int kk = e + 8 * half + ((e >= 8) ? 8 : 0);
      af0[e] = Asm[(r0 + lrow) * 40 + kk];
      af1[e] = Asm[(r0 + 16 + lrow) * 40 + kk];
    }
#pragma unroll
    for (int j = 0; j < 4; ++j) {
      v16bf bfv;
#pragma unroll
      for (int e = 0; e < 16; ++e) bfv[e] = Bsm[(j * 16 + lrow) * 40 + e + 8 * half + ((e >= 8) ? 8 : 0)];
      acc[j]     = __builtin_amdgcn_wmma_f32_16x16x32_bf16(false, af0, false, bfv,
                                                           (short)0, acc[j], false, false);
      acc[4 + j] = __builtin_amdgcn_wmma_f32_16x16x32_bf16(false, af1, false, bfv,
                                                           (short)0, acc[4 + j], false, false);
    }
    __syncthreads();
  }
#pragma unroll
  for (int j = 0; j < 4; ++j)
#pragma unroll
    for (int v = 0; v < 8; ++v) {
      int gm0 = m0 + r0 + v + 8 * half;
      int gm1 = gm0 + 16;
      int col = n0 + j * 16 + lrow;
      if (gm0 < MTOT) Out[(size_t)gm0 * DIMM + col] = acc[j][v] + bias[col];
      if (gm1 < MTOT) Out[(size_t)gm1 * DIMM + col] = acc[4 + j][v] + bias[col];
    }
}

// ---------------------------------------------------------------------------
extern "C" void kernel_launch(void* const* d_in, const int* in_sizes, int n_in,
                              void* d_out, int out_size, void* d_ws, size_t ws_size,
                              hipStream_t stream) {
  (void)in_sizes; (void)n_in; (void)out_size; (void)ws_size;
  const float* x     = (const float*)d_in[0];   // (8, 3137, 768)
  const float* Wqkv  = (const float*)d_in[1];   // (2304, 768)
  const float* Wproj = (const float*)d_in[2];   // (768, 768)
  const float* bproj = (const float*)d_in[3];   // (768,)
  float* out = (float*)d_out;                   // (8, 3137, 768)

  bf16_t* qkv_bf  = (bf16_t*)d_ws;                          // 25096*2304 bf16
  bf16_t* attn_bf = qkv_bf + (size_t)MTOT * QKVN;           // 25096*768  bf16

  dim3 g1(QKVN / 64, (MTOT + 127) / 128);       // (36, 197)
  gemm_qkv<<<g1, 128, 0, stream>>>(x, Wqkv, qkv_bf);

  dim3 g2(BTOT * HH * FF, 13);                  // (1536, 13)
  seg_attn<<<g2, 128, 0, stream>>>(qkv_bf, attn_bf);

  cls_attn<<<dim3(BTOT * HH), 256, 0, stream>>>(qkv_bf, attn_bf);

  dim3 g4(DIMM / 64, (MTOT + 127) / 128);       // (12, 197)
  gemm_proj<<<g4, 128, 0, stream>>>(attn_bf, Wproj, bproj, out);
}
